// TransformerDecoderLayer_9371618640157
// MI455X (gfx1250) — matrix operand order repaired and checked by reading, not compiled, not run on hardware
//
#include <hip/hip_runtime.h>
#include <math.h>

// ---------------------------------------------------------------------------
// Types for CDNA5 WMMA (wave32): v_wmma_f32_16x16x32_bf16
// ---------------------------------------------------------------------------
typedef __bf16 bf16;
typedef __bf16 v16bf __attribute__((ext_vector_type(16)));
typedef __bf16 v8bf  __attribute__((ext_vector_type(8)));
typedef float  v8f   __attribute__((ext_vector_type(8)));
typedef float  v4f   __attribute__((ext_vector_type(4)));

#define BDIM 256   // 8 waves of 32
#define SEQ  1024
#define DMODEL 1024
#define NHEAD 16
#define DHEAD 64
#define BATCH 4
#define MROWS (BATCH * SEQ)   // 4096 flattened rows
#define FFDIM 4096

// Fragment element mappings (16-bit operands, ISA 7.12.2):
//  A (16x32, MxK): lane l, elem j -> M = l%16, K = 16*(j/8) + 8*(l/16) + (j%8)
//     => two contiguous 8-element K runs at offsets 8*(l/16) and 16+8*(l/16).
//  B (32x16, KxN): lane l, elem j -> N = l%16, K = 16*(l/16) + j here, A too (ISA B order = A's above)
//     => one contiguous 16-element K run at offset 16*(l/16).
//  C/D (16x16 f32): VGPR r -> M = r + 8*(l/16), N = l%16.

// A-fragment from f32: 16 contiguous floats at p
__device__ __forceinline__ v16bf a_frag_f32(const float* p) {
    v4f a0 = *(const v4f*)(p);
    v4f a1 = *(const v4f*)(p + 4);
    v4f b0 = *(const v4f*)(p + 8);
    v4f b1 = *(const v4f*)(p + 12);
    v16bf r;
#pragma unroll
    for (int i = 0; i < 4; ++i) {
        r[i]      = (bf16)a0[i];
        r[4 + i]  = (bf16)a1[i];
        r[8 + i]  = (bf16)b0[i];
        r[12 + i] = (bf16)b1[i];
    }
    return r;
}

// A-fragment from bf16 LDS: 16 contiguous at p (both halves 16B aligned)
__device__ __forceinline__ v16bf a_frag_lds(const bf16* p) {
    v8bf lo = *(const v8bf*)(p);
    v8bf hi = *(const v8bf*)(p + 8);
    return __builtin_shufflevector(lo, hi, 0, 1, 2, 3, 4, 5, 6, 7,
                                   8, 9, 10, 11, 12, 13, 14, 15);
}

// B-fragment from f32: 16 contiguous floats at p
__device__ __forceinline__ v16bf b_frag_f32(const float* p) {
    v4f a0 = *(const v4f*)(p);
    v4f a1 = *(const v4f*)(p + 4);
    v4f a2 = *(const v4f*)(p + 8);
    v4f a3 = *(const v4f*)(p + 12);
    v16bf r;
#pragma unroll
    for (int i = 0; i < 4; ++i) {
        r[i]      = (bf16)a0[i];
        r[4 + i]  = (bf16)a1[i];
        r[8 + i]  = (bf16)a2[i];
        r[12 + i] = (bf16)a3[i];
    }
    return r;
}

// B-fragment from bf16 LDS: 16 contiguous at p (32B, 16B aligned)
__device__ __forceinline__ v16bf b_frag_lds(const bf16* p) {
    v8bf lo = *(const v8bf*)(p);
    v8bf hi = *(const v8bf*)(p + 8);
    return __builtin_shufflevector(lo, hi, 0, 1, 2, 3, 4, 5, 6, 7,
                                   8, 9, 10, 11, 12, 13, 14, 15);
}

__device__ __forceinline__ v8f wmma_bf16(v16bf a, v16bf b, v8f c) {
    return __builtin_amdgcn_wmma_f32_16x16x32_bf16(
        /*neg_a=*/false, a, /*neg_b=*/false, b,
        /*c_mod=*/(short)0, c, /*reuse_a=*/false, /*reuse_b=*/false);
}

// ---------------------------------------------------------------------------
// GEMM: C[M,N] = act( A[M,K] @ W[K,N] + bias [+ residual] )
// Block tile 128x128, K-chunk 32, bf16 operands staged in LDS, f32 accum.
// 8 waves: wave (wy,wx) with wy=0..3, wx=0..1 owns rows wy*32..+31, cols wx*64..+63
// ---------------------------------------------------------------------------
#define LDS_STRIDE 40  // bf16 elements per row (80B: keeps 16B alignment, spreads banks)

__global__ __launch_bounds__(BDIM) void gemm_bias_kernel(
    const float* __restrict__ A, const float* __restrict__ W,
    const float* __restrict__ bias, const float* __restrict__ residual,
    float* __restrict__ C, int M, int N, int K, int relu)
{
    __shared__ bf16 As[128 * LDS_STRIDE];
    __shared__ bf16 Bs[128 * LDS_STRIDE];

    const int tid  = threadIdx.x;
    const int w    = tid >> 5;
    const int lane = tid & 31;
    const int lc   = lane & 15;   // col/row within 16
    const int lh   = lane >> 4;   // half select
    const int wy   = w >> 1;
    const int wx   = w & 1;
    const int m0   = blockIdx.y * 128;
    const int n0   = blockIdx.x * 128;

    // staging assignments
    const int arow = tid >> 1;          // 0..127
    const int acb  = (tid & 1) * 16;    // 0 or 16
    const int bkk  = tid >> 3;          // 0..31
    const int bnb  = (tid & 7) * 16;    // 0..112

    v8f acc[2][4] = {};

    for (int k0 = 0; k0 < K; k0 += 32) {
        // ---- stage A tile (128x32 f32 -> bf16 LDS, row-major) ----
        {
            const float* src = A + (size_t)(m0 + arow) * K + k0 + acb;
            if (k0 + 32 < K) __builtin_prefetch(src + 32, 0, 1);
            v4f f0 = *(const v4f*)(src);
            v4f f1 = *(const v4f*)(src + 4);
            v4f f2 = *(const v4f*)(src + 8);
            v4f f3 = *(const v4f*)(src + 12);
            v8bf lo, hi;
#pragma unroll
            for (int i = 0; i < 4; ++i) {
                lo[i]     = (bf16)f0[i];
                lo[4 + i] = (bf16)f1[i];
                hi[i]     = (bf16)f2[i];
                hi[4 + i] = (bf16)f3[i];
            }
            *(v8bf*)&As[arow * LDS_STRIDE + acb]     = lo;
            *(v8bf*)&As[arow * LDS_STRIDE + acb + 8] = hi;
        }
        // ---- stage W tile transposed (32x128 f32 -> Bs[n][k] bf16) ----
        {
            const float* src = W + (size_t)(k0 + bkk) * N + n0 + bnb;
            if (k0 + 32 < K) __builtin_prefetch(src + (size_t)32 * N, 0, 1);
            v4f g0 = *(const v4f*)(src);
            v4f g1 = *(const v4f*)(src + 4);
            v4f g2 = *(const v4f*)(src + 8);
            v4f g3 = *(const v4f*)(src + 12);
#pragma unroll
            for (int i = 0; i < 4; ++i) {
                Bs[(bnb + i)      * LDS_STRIDE + bkk] = (bf16)g0[i];
                Bs[(bnb + 4 + i)  * LDS_STRIDE + bkk] = (bf16)g1[i];
                Bs[(bnb + 8 + i)  * LDS_STRIDE + bkk] = (bf16)g2[i];
                Bs[(bnb + 12 + i) * LDS_STRIDE + bkk] = (bf16)g3[i];
            }
        }
        __syncthreads();

        v16bf af[2], bfm[4];
#pragma unroll
        for (int mt = 0; mt < 2; ++mt)
            af[mt] = a_frag_lds(&As[(wy * 32 + mt * 16 + lc) * LDS_STRIDE + 16 * lh]);
#pragma unroll
        for (int nt = 0; nt < 4; ++nt)
            bfm[nt] = b_frag_lds(&Bs[(wx * 64 + nt * 16 + lc) * LDS_STRIDE + 16 * lh]);

#pragma unroll
        for (int mt = 0; mt < 2; ++mt)
#pragma unroll
            for (int nt = 0; nt < 4; ++nt)
                acc[mt][nt] = wmma_bf16(af[mt], bfm[nt], acc[mt][nt]);

        __syncthreads();
    }

    // ---- epilogue: bias, residual, relu, store ----
#pragma unroll
    for (int mt = 0; mt < 2; ++mt) {
#pragma unroll
        for (int nt = 0; nt < 4; ++nt) {
            const int col = n0 + wx * 64 + nt * 16 + lc;
            const float bv = bias[col];
            const int rbase = m0 + wy * 32 + mt * 16 + 8 * lh;
#pragma unroll
            for (int r = 0; r < 8; ++r) {
                const int row = rbase + r;
                float v = acc[mt][nt][r] + bv;
                if (residual) v += residual[(size_t)row * N + col];
                if (relu) v = fmaxf(v, 0.0f);
                C[(size_t)row * N + col] = v;
            }
        }
    }
}

// ---------------------------------------------------------------------------
// Attention: per block = one (b, h, 64-query tile).
// Dynamic LDS: scores f32[64][1024] (256KB) + Vt bf16[64][40] chunk (5KB).
// Pass1: S = QK^T/8 + mask   (WMMA, fragments straight from global f32)
// Pass2: row softmax (wave32 shfl reductions)
// Pass3: O = P @ V           (WMMA, V chunk transposed through LDS)
// ---------------------------------------------------------------------------
#define ATTN_SMEM_BYTES (64 * SEQ * 4 + 64 * LDS_STRIDE * 2)

__global__ __launch_bounds__(BDIM) void attention_kernel(
    const float* __restrict__ Q, const float* __restrict__ Km,
    const float* __restrict__ Vm, const int* __restrict__ mask,
    float* __restrict__ O)
{
    extern __shared__ char raw[];
    float* sc = (float*)raw;                       // [64][1024]
    bf16*  vt = (bf16*)(raw + 64 * SEQ * 4);       // [64][LDS_STRIDE]

    const int tid  = threadIdx.x;
    const int w    = tid >> 5;
    const int lane = tid & 31;
    const int lc   = lane & 15;
    const int lh   = lane >> 4;

    const int qb = blockIdx.x & 15;          // query tile within sequence
    const int h  = (blockIdx.x >> 4) & 15;   // head
    const int b  = blockIdx.x >> 8;          // batch
    const int base_q = qb * 64;

    // ---------------- Pass 1: scores ----------------
    {
        const int mt    = w & 3;     // 16-row tile within the 64 queries
        const int ntoff = w >> 2;    // 0/1 -> even/odd key tiles

        v16bf aq[2];
#pragma unroll
        for (int ks = 0; ks < 2; ++ks) {
            const float* qp = Q + ((size_t)(b * SEQ + base_q + mt * 16 + lc)) * DMODEL
                                + h * DHEAD + ks * 32 + 16 * lh;
            aq[ks] = a_frag_f32(qp);
        }

        for (int j = 0; j < 32; ++j) {
            const int nt = ntoff + 2 * j;       // key tile 0..63
            v8f acc = {};
#pragma unroll
            for (int ks = 0; ks < 2; ++ks) {
                const float* kp = Km + ((size_t)(b * SEQ + nt * 16 + lc)) * DMODEL
                                     + h * DHEAD + ks * 32 + 16 * lh;
                v16bf bk = b_frag_f32(kp);
                acc = wmma_bf16(aq[ks], bk, acc);
            }
            const int sk = nt * 16 + lc;
#pragma unroll
            for (int r = 0; r < 8; ++r) {
                const int sql = mt * 16 + r + 8 * lh;     // local query row 0..63
                const int mv  = mask[((size_t)b * SEQ + base_q + sql) * SEQ + sk];
                float v = acc[r] * 0.125f + (1.0f - (float)mv) * (-1e9f);
                sc[sql * SEQ + sk] = v;
            }
        }
    }
    __syncthreads();

    // ---------------- Pass 2: softmax ----------------
    {
#pragma unroll
        for (int rl = 0; rl < 8; ++rl) {
            float* rp = sc + (w * 8 + rl) * SEQ;
            float mx = -__builtin_inff();
            for (int c = lane; c < SEQ; c += 32) mx = fmaxf(mx, rp[c]);
#pragma unroll
            for (int off = 16; off >= 1; off >>= 1)
                mx = fmaxf(mx, __shfl_xor(mx, off, 32));
            float sum = 0.0f;
            for (int c = lane; c < SEQ; c += 32) {
                float e = __expf(rp[c] - mx);
                rp[c] = e;
                sum += e;
            }
#pragma unroll
            for (int off = 16; off >= 1; off >>= 1)
                sum += __shfl_xor(sum, off, 32);
            const float inv = 1.0f / sum;
            for (int c = lane; c < SEQ; c += 32) rp[c] *= inv;
        }
    }
    __syncthreads();

    // ---------------- Pass 3: O = P @ V ----------------
    {
        const int mt  = w & 3;
        const int nt0 = (w >> 2) * 2;     // two 16-col output tiles per wave
        v8f oacc[2] = {};

        for (int kk = 0; kk < 32; ++kk) {
            // stage V chunk transposed: vt[d][s] = V[kk*32+s][h*64+d]
            {
                const int s  = kk * 32 + (tid & 31);
                const int db = (tid >> 5) * 8;
                const float* vp = Vm + ((size_t)(b * SEQ + s)) * DMODEL + h * DHEAD + db;
                v4f f0 = *(const v4f*)(vp);
                v4f f1 = *(const v4f*)(vp + 4);
#pragma unroll
                for (int i = 0; i < 4; ++i) {
                    vt[(db + i)     * LDS_STRIDE + (tid & 31)] = (bf16)f0[i];
                    vt[(db + 4 + i) * LDS_STRIDE + (tid & 31)] = (bf16)f1[i];
                }
            }
            __syncthreads();

            const float* pp = sc + (mt * 16 + lc) * SEQ + kk * 32 + 16 * lh;
            v16bf ap = a_frag_f32(pp);
#pragma unroll
            for (int j = 0; j < 2; ++j) {
                v16bf bv = b_frag_lds(&vt[((nt0 + j) * 16 + lc) * LDS_STRIDE + 16 * lh]);
                oacc[j] = wmma_bf16(ap, bv, oacc[j]);
            }
            __syncthreads();
        }

#pragma unroll
        for (int j = 0; j < 2; ++j) {
            const int d = (nt0 + j) * 16 + lc;
#pragma unroll
            for (int r = 0; r < 8; ++r) {
                const int sq = base_q + mt * 16 + r + 8 * lh;
                O[((size_t)(b * SEQ + sq)) * DMODEL + h * DHEAD + d] = oacc[j][r];
            }
        }
    }
}

// ---------------------------------------------------------------------------
// LayerNorm over last dim (1024). One block (256 threads) per row.
// ---------------------------------------------------------------------------
__global__ __launch_bounds__(BDIM) void layernorm_kernel(
    const float* __restrict__ X, const float* __restrict__ g,
    const float* __restrict__ be, float* __restrict__ Y)
{
    __shared__ float ssum[8], ssq[8];
    const int row  = blockIdx.x;
    const int tid  = threadIdx.x;
    const int w    = tid >> 5;
    const int lane = tid & 31;

    float xv[4];
    float s = 0.0f, q2 = 0.0f;
#pragma unroll
    for (int k = 0; k < 4; ++k) {
        const int c = tid + 256 * k;
        float v = X[(size_t)row * DMODEL + c];
        xv[k] = v;
        s += v;
        q2 += v * v;
    }
#pragma unroll
    for (int off = 16; off >= 1; off >>= 1) {
        s  += __shfl_xor(s, off, 32);
        q2 += __shfl_xor(q2, off, 32);
    }
    if (lane == 0) { ssum[w] = s; ssq[w] = q2; }
    __syncthreads();
    float ts = 0.0f, tq = 0.0f;
#pragma unroll
    for (int i = 0; i < 8; ++i) { ts += ssum[i]; tq += ssq[i]; }
    const float mean = ts * (1.0f / DMODEL);
    const float var  = tq * (1.0f / DMODEL) - mean * mean;
    const float rinv = rsqrtf(var + 1e-5f);
#pragma unroll
    for (int k = 0; k < 4; ++k) {
        const int c = tid + 256 * k;
        Y[(size_t)row * DMODEL + c] = (xv[k] - mean) * rinv * g[c] + be[c];
    }
}

// ---------------------------------------------------------------------------
// Host orchestration
// ---------------------------------------------------------------------------
extern "C" void kernel_launch(void* const* d_in, const int* in_sizes, int n_in,
                              void* d_out, int out_size, void* d_ws, size_t ws_size,
                              hipStream_t stream) {
    const float* tgt    = (const float*)d_in[0];
    const float* memry  = (const float*)d_in[1];
    const int*   mask   = (const int*)  d_in[2];
    const float* sa_wq  = (const float*)d_in[3];
    const float* sa_bq  = (const float*)d_in[4];
    const float* sa_wk  = (const float*)d_in[5];
    const float* sa_bk  = (const float*)d_in[6];
    const float* sa_wv  = (const float*)d_in[7];
    const float* sa_bv  = (const float*)d_in[8];
    const float* sa_wo  = (const float*)d_in[9];
    const float* sa_bo  = (const float*)d_in[10];
    const float* ca_wq  = (const float*)d_in[11];
    const float* ca_bq  = (const float*)d_in[12];
    const float* ca_wk  = (const float*)d_in[13];
    const float* ca_bk  = (const float*)d_in[14];
    const float* ca_wv  = (const float*)d_in[15];
    const float* ca_bv  = (const float*)d_in[16];
    const float* ca_wo  = (const float*)d_in[17];
    const float* ca_bo  = (const float*)d_in[18];
    const float* w1     = (const float*)d_in[19];
    const float* b1     = (const float*)d_in[20];
    const float* w2     = (const float*)d_in[21];
    const float* b2     = (const float*)d_in[22];
    const float* g1     = (const float*)d_in[23];
    const float* be1    = (const float*)d_in[24];
    const float* g2     = (const float*)d_in[25];
    const float* be2    = (const float*)d_in[26];
    const float* g3     = (const float*)d_in[27];
    const float* be3    = (const float*)d_in[28];
    (void)in_sizes; (void)n_in; (void)out_size; (void)ws_size;

    // Workspace carve-out (f32): six 16MB buffers + one 64MB FF buffer
    const size_t CH = (size_t)MROWS * DMODEL;          // 4M floats
    float* ws   = (float*)d_ws;
    float* tb   = ws;            // residual stream
    float* qb   = ws + 1 * CH;   // Q / later reused as t2
    float* kb   = ws + 2 * CH;
    float* vb   = ws + 3 * CH;
    float* ab   = ws + 4 * CH;   // attention output
    float* pb   = ws + 5 * CH;   // projection output (pre-LN)
    float* ffb  = ws + 6 * CH;   // [4096 x 4096]

    const dim3 blk(BDIM);
    const dim3 gD (DMODEL / 128, MROWS / 128);   // N=1024 GEMMs
    const dim3 gFF(FFDIM  / 128, MROWS / 128);   // N=4096 GEMM
    const dim3 gAttn(BATCH * NHEAD * (SEQ / 64));
    const dim3 gLN(MROWS);

    // ---- self-attention ----
    gemm_bias_kernel<<<gD, blk, 0, stream>>>(tgt, sa_wq, sa_bq, nullptr, qb,
                                             MROWS, DMODEL, DMODEL, 0);
    gemm_bias_kernel<<<gD, blk, 0, stream>>>(tgt, sa_wk, sa_bk, nullptr, kb,
                                             MROWS, DMODEL, DMODEL, 0);
    gemm_bias_kernel<<<gD, blk, 0, stream>>>(tgt, sa_wv, sa_bv, nullptr, vb,
                                             MROWS, DMODEL, DMODEL, 0);
    attention_kernel<<<gAttn, blk, ATTN_SMEM_BYTES, stream>>>(qb, kb, vb, mask, ab);
    gemm_bias_kernel<<<gD, blk, 0, stream>>>(ab, sa_wo, sa_bo, tgt, pb,
                                             MROWS, DMODEL, DMODEL, 0);
    layernorm_kernel<<<gLN, blk, 0, stream>>>(pb, g1, be1, tb);

    // ---- cross-attention ----
    gemm_bias_kernel<<<gD, blk, 0, stream>>>(tb,    ca_wq, ca_bq, nullptr, qb,
                                             MROWS, DMODEL, DMODEL, 0);
    gemm_bias_kernel<<<gD, blk, 0, stream>>>(memry, ca_wk, ca_bk, nullptr, kb,
                                             MROWS, DMODEL, DMODEL, 0);
    gemm_bias_kernel<<<gD, blk, 0, stream>>>(memry, ca_wv, ca_bv, nullptr, vb,
                                             MROWS, DMODEL, DMODEL, 0);
    attention_kernel<<<gAttn, blk, ATTN_SMEM_BYTES, stream>>>(qb, kb, vb, mask, ab);
    gemm_bias_kernel<<<gD, blk, 0, stream>>>(ab, ca_wo, ca_bo, tb, pb,
                                             MROWS, DMODEL, DMODEL, 0);
    layernorm_kernel<<<gLN, blk, 0, stream>>>(pb, g2, be2, qb);  // qb = t2

    // ---- FFN ----
    gemm_bias_kernel<<<gFF, blk, 0, stream>>>(qb, w1, b1, nullptr, ffb,
                                              MROWS, FFDIM, DMODEL, 1 /*relu*/);
    gemm_bias_kernel<<<gD, blk, 0, stream>>>(ffb, w2, b2, qb, pb,
                                             MROWS, DMODEL, FFDIM, 0);
    layernorm_kernel<<<gLN, blk, 0, stream>>>(pb, g3, be3, (float*)d_out);
}
